// Attention_76184129896530
// MI455X (gfx1250) — matrix operand order repaired and checked by reading, not compiled, not run on hardware
//
#include <hip/hip_runtime.h>
#include <hip/hip_bf16.h>
#include <math.h>

// ---------------- problem constants ----------------
#define HEADS 8
#define DHEAD 64
#define LMARK 256
#define SEQN  8192
#define BATCH 2
#define BH    (BATCH * HEADS)     // 16
#define DIM   512
#define INNER 512
#define QKVC  1536
#define ROWS  (BATCH * SEQN)      // 16384
#define KCONV 33
#define PINV_ITERS 6
#define KSPLIT 16                 // split-K for attn3@v

// ---------------- WMMA plumbing (gfx1250, wave32) ----------------
typedef __attribute__((ext_vector_type(16))) _Float16 v16h;
typedef __attribute__((ext_vector_type(8)))  float    v8f;

union F16Frag { v16h v; _Float16 h[16]; uint4 q[2]; };
union F32Acc  { v8f  v; float    f[8]; };

__device__ __forceinline__ v8f wmma_f16(v16h a, v16h b, v8f c) {
  // D = A(16x32) * B(32x16) + C, f32 accumulate
  return __builtin_amdgcn_wmma_f32_16x16x32_f16(false, a, false, b, (short)0, c,
                                                false, false);
}

// A fragment 16x32 from row-major [..][ld] f16, `base` -> (row0, k0).
// K order used here (same in load_bt): half-wave hi holds the contiguous K run {16*hi .. 16*hi+15}.
__device__ __forceinline__ v16h load_a(const _Float16* __restrict__ base, int ld, int lane) {
  const _Float16* p = base + (size_t)(lane & 15) * ld + ((lane >> 4) << 4);
  F16Frag f;
  f.q[0] = *(const uint4*)(p);
  f.q[1] = *(const uint4*)(p + 8);
  return f.v;
}

// B fragment 32x16 where B is stored TRANSPOSED: Bt row-major [N][K], base -> (n0, k0).
// B k order used here (same as load_a): half-wave hi holds the contiguous K run {16*hi .. 16*hi+15} of column n.
__device__ __forceinline__ v16h load_bt(const _Float16* __restrict__ base, int ld, int lane) {
  const _Float16* p = base + (size_t)(lane & 15) * ld + ((lane >> 4) << 4);
  F16Frag f;
  f.q[0] = *(const uint4*)(p);
  f.q[1] = *(const uint4*)(p + 8);
  return f.v;
}

// ---------------- small prep kernels ----------------
__global__ void k_cvt_half(const float* __restrict__ in, _Float16* __restrict__ out, int n) {
  for (int i = blockIdx.x * blockDim.x + threadIdx.x; i < n; i += gridDim.x * blockDim.x)
    out[i] = (_Float16)in[i];
}

// W_qkv [512][1536] -> transposed f16 [1536][512]; fold q-scale d^-0.5 into q columns.
__global__ void k_prep_wqkv(const float* __restrict__ W, _Float16* __restrict__ Wt) {
  const int n = QKVC * DIM;
  for (int i = blockIdx.x * blockDim.x + threadIdx.x; i < n; i += gridDim.x * blockDim.x) {
    int c = i / DIM, k = i - c * DIM;
    float v = W[(size_t)k * QKVC + c];
    if (c < DIM) v *= 0.125f;  // 64^-0.5
    Wt[(size_t)c * DIM + k] = (_Float16)v;
  }
}

// W_out [512][512] -> transposed f16 [512][512]
__global__ void k_prep_wout(const float* __restrict__ W, _Float16* __restrict__ Wt) {
  const int n = DIM * INNER;
  for (int i = blockIdx.x * blockDim.x + threadIdx.x; i < n; i += gridDim.x * blockDim.x) {
    int c = i / INNER, k = i - c * INNER;
    Wt[(size_t)c * INNER + k] = (_Float16)W[(size_t)k * DIM + c];
  }
}

// ---------------- QKV projection: [16384,512] x [512,1536], 32x64 tile/wave --------
__global__ void k_gemm_qkv(const _Float16* __restrict__ xh, const _Float16* __restrict__ wt,
                           _Float16* __restrict__ qh, _Float16* __restrict__ kh,
                           _Float16* __restrict__ vh) {
  int lane = threadIdx.x & 31;
  int wav  = blockIdx.x * (blockDim.x >> 5) + (threadIdx.x >> 5);   // 12288 waves
  int mt = wav / 24, nt = wav - mt * 24;        // 512 strips x 24 col-tiles
  int m0 = mt << 5, n0 = nt << 6;
  F32Acc c[2][4];
  v8f zero = {0.f, 0.f, 0.f, 0.f, 0.f, 0.f, 0.f, 0.f};
#pragma unroll
  for (int u = 0; u < 2; ++u)
#pragma unroll
    for (int j = 0; j < 4; ++j) c[u][j].v = zero;

  for (int kk = 0; kk < DIM; kk += 32) {
    v16h a0 = load_a(xh + (size_t)m0 * DIM + kk, DIM, lane);
    v16h a1 = load_a(xh + (size_t)(m0 + 16) * DIM + kk, DIM, lane);
#pragma unroll
    for (int j = 0; j < 4; ++j) {
      v16h b = load_bt(wt + (size_t)(n0 + j * 16) * DIM + kk, DIM, lane);
      c[0][j].v = wmma_f16(a0, b, c[0][j].v);
      c[1][j].v = wmma_f16(a1, b, c[1][j].v);
    }
  }
  int hi = lane >> 4, cl = lane & 15;
#pragma unroll
  for (int u = 0; u < 2; ++u)
#pragma unroll
    for (int j = 0; j < 4; ++j)
#pragma unroll
      for (int r = 0; r < 8; ++r) {
        int row = m0 + u * 16 + r + 8 * hi;
        int col = n0 + j * 16 + cl;
        int part = col >> 9, rem = col & 511;
        int h = rem >> 6, dd = rem & 63;
        int bb = row >> 13, nn = row & 8191;
        _Float16* dst = (part == 0) ? qh : ((part == 1) ? kh : vh);
        dst[(((size_t)(bb * HEADS + h)) * SEQN + nn) * DHEAD + dd] = (_Float16)c[u][j].f[r];
      }
}

// ---------------- landmark means (l = 32) ----------------
__global__ void k_landmarks(const _Float16* __restrict__ qh, const _Float16* __restrict__ kh,
                            _Float16* __restrict__ ql, _Float16* __restrict__ kl) {
  const int n = BH * LMARK * DHEAD;
  for (int i = blockIdx.x * blockDim.x + threadIdx.x; i < n; i += gridDim.x * blockDim.x) {
    int bh = i / (LMARK * DHEAD);
    int rem = i - bh * (LMARK * DHEAD);
    int mi = rem / DHEAD, dd = rem - mi * DHEAD;
    size_t base = ((size_t)bh * SEQN + (size_t)mi * 32) * DHEAD + dd;
    float sq = 0.f, sk = 0.f;
#pragma unroll 8
    for (int t = 0; t < 32; ++t) {
      sq += (float)qh[base + (size_t)t * DHEAD];
      sk += (float)kh[base + (size_t)t * DHEAD];
    }
    ql[i] = (_Float16)(sq * (1.f / 32.f));
    kl[i] = (_Float16)(sk * (1.f / 32.f));
  }
}

// ---------------- fused GEMM + row-softmax over 256 landmark columns ----------------
// OUT[M][256] = softmax_row( Q[M][64] @ KL[256][64]^T ) per (b,h).
// One wave: 16 rows x 256 cols. KL tile (32 KB) staged in LDS via the gfx1250
// async-to-LDS path (GLOBAL_LOAD_ASYNC_TO_LDS_B128 + ASYNCcnt), shared by 4 waves.
__global__ void k_qk_softmax(const _Float16* __restrict__ Q, const _Float16* __restrict__ KL,
                             _Float16* __restrict__ OUT, int M) {
  __shared__ _Float16 sB[LMARK * DHEAD];     // 32 KB of 320 KB/WGP
  int lane = threadIdx.x & 31;
  int wav  = blockIdx.x * (blockDim.x >> 5) + (threadIdx.x >> 5);
  int strips = M >> 4;                       // strips % 4 == 0 -> bh uniform per block
  int bh = wav / strips, mt = wav - bh * strips;
  int m0 = mt << 4;

  // async stage of KL[bh] into LDS: each lane copies 16B per issue, no VGPR bounce.
  // Flat shared-aperture address low 32 bits == workgroup-relative LDS byte offset.
  {
    const _Float16* src = KL + (size_t)bh * LMARK * DHEAD;
    unsigned lbase = (unsigned)(size_t)(&sB[0]);
    for (int i = threadIdx.x; i < (LMARK * DHEAD) / 8; i += blockDim.x) {
      unsigned ldst = lbase + (unsigned)i * 16u;
      const _Float16* g = src + (size_t)i * 8;
      asm volatile("global_load_async_to_lds_b128 %0, %1, off"
                   :: "v"(ldst), "v"(g) : "memory");
    }
    asm volatile("s_wait_asynccnt 0x0" ::: "memory");
  }
  __syncthreads();

  const _Float16* A = Q + ((size_t)bh * M + m0) * DHEAD;
  F32Acc c[16];
  v8f zero = {0.f, 0.f, 0.f, 0.f, 0.f, 0.f, 0.f, 0.f};
#pragma unroll
  for (int t = 0; t < 16; ++t) c[t].v = zero;

  for (int kk = 0; kk < DHEAD; kk += 32) {
    v16h a = load_a(A + kk, DHEAD, lane);
#pragma unroll
    for (int t = 0; t < 16; ++t) {
      v16h b = load_bt(sB + (size_t)(t * 16) * DHEAD + kk, DHEAD, lane);
      c[t].v = wmma_f16(a, b, c[t].v);
    }
  }

  int hi = lane >> 4, cl = lane & 15;
#pragma unroll
  for (int r = 0; r < 8; ++r) {
    float mx = -1e30f;
#pragma unroll
    for (int t = 0; t < 16; ++t) mx = fmaxf(mx, c[t].f[r]);
    // reduce across the 16-lane half (xor masks 1,2,4,8 stay within the half)
#pragma unroll
    for (int off = 1; off < 16; off <<= 1) mx = fmaxf(mx, __shfl_xor(mx, off, 32));
    float s = 0.f;
#pragma unroll
    for (int t = 0; t < 16; ++t) { float e = __expf(c[t].f[r] - mx); c[t].f[r] = e; s += e; }
#pragma unroll
    for (int off = 1; off < 16; off <<= 1) s += __shfl_xor(s, off, 32);
    float inv = 1.f / s;
    int row = m0 + r + 8 * hi;
    _Float16* o = OUT + ((size_t)bh * M + row) * LMARK + cl;
#pragma unroll
    for (int t = 0; t < 16; ++t) o[t * 16] = (_Float16)(c[t].f[r] * inv);
  }
}

// ---------------- pinv init: global max of column sums of attn2 ----------------
__global__ void k_zero_u32(unsigned* p) { if (blockIdx.x == 0 && threadIdx.x == 0) *p = 0u; }

__global__ void k_colsum_max(const _Float16* __restrict__ a2, unsigned* __restrict__ scale) {
  int i = blockIdx.x * blockDim.x + threadIdx.x;      // 4096 = BH*256 columns
  if (i >= BH * LMARK) return;
  int bh = i >> 8, col = i & 255;
  const _Float16* p = a2 + ((size_t)bh << 16) + col;
  float s = 0.f;
  for (int r = 0; r < LMARK; ++r) s += (float)p[(size_t)r * LMARK];
  atomicMax(scale, __float_as_uint(s));               // all values >= 0
}

// z0 = attn2^T / S (row-sums of softmax are exactly 1 -> S = max colsum).
// Emit both natural z0 and transposed z0T (= attn2 / S) so every later GEMM
// reads its B operand through the contiguous load_bt path.
__global__ void k_init_z(const _Float16* __restrict__ a2, _Float16* __restrict__ z,
                         _Float16* __restrict__ zT, const unsigned* __restrict__ scale) {
  float inv = 1.f / __uint_as_float(*scale);
  const int n = BH * LMARK * LMARK;
  for (int i = blockIdx.x * blockDim.x + threadIdx.x; i < n; i += gridDim.x * blockDim.x) {
    int bh = i >> 16, rem = i & 65535;
    int r = rem >> 8, cc = rem & 255;
    z[i]  = (_Float16)((float)a2[((size_t)bh << 16) + ((size_t)cc << 8) + r] * inv);
    zT[i] = (_Float16)((float)a2[i] * inv);
  }
}

// pT = (7I - xz)^T  (only the transposed form is ever consumed)
__global__ void k_7i_minus_T(const _Float16* __restrict__ xz, _Float16* __restrict__ pT) {
  const int n = BH * LMARK * LMARK;
  for (int i = blockIdx.x * blockDim.x + threadIdx.x; i < n; i += gridDim.x * blockDim.x) {
    int bh = i >> 16, rem = i & 65535;
    int cc = rem >> 8, r = rem & 255;   // output index [bh][cc][r]
    float v = ((r == cc) ? 7.f : 0.f) - (float)xz[((size_t)bh << 16) + ((size_t)r << 8) + cc];
    pT[i] = (_Float16)v;
  }
}

// ---------------- batched 256^3 GEMM: OUT{n,t} = diag*I + scale*(A@B) ----------------
// A natural [K-major rows]; B supplied TRANSPOSED (Bt[N][K]) -> contiguous b128 loads.
// Optionally stores natural and/or transposed result (epilogue scatter is free).
__global__ void k_gemm256(const _Float16* __restrict__ A, const _Float16* __restrict__ Bt,
                          _Float16* __restrict__ OUTn, _Float16* __restrict__ OUTt,
                          float diag, float scale) {
  int lane = threadIdx.x & 31;
  int wav  = blockIdx.x * (blockDim.x >> 5) + (threadIdx.x >> 5);   // 1024 waves
  int bh = wav >> 6, t = wav & 63;
  int m0 = (t >> 2) << 4, n0 = (t & 3) << 6;
  const _Float16* Ab = A + ((size_t)bh << 16) + (size_t)m0 * 256;
  const _Float16* Bb = Bt + ((size_t)bh << 16);
  F32Acc c[4];
  v8f zero = {0.f, 0.f, 0.f, 0.f, 0.f, 0.f, 0.f, 0.f};
#pragma unroll
  for (int j = 0; j < 4; ++j) c[j].v = zero;

  for (int kk = 0; kk < 256; kk += 32) {
    v16h a = load_a(Ab + kk, 256, lane);
#pragma unroll
    for (int j = 0; j < 4; ++j) {
      v16h b = load_bt(Bb + (size_t)(n0 + j * 16) * 256 + kk, 256, lane);
      c[j].v = wmma_f16(a, b, c[j].v);
    }
  }
  int hi = lane >> 4, cl = lane & 15;
#pragma unroll
  for (int j = 0; j < 4; ++j)
#pragma unroll
    for (int r = 0; r < 8; ++r) {
      int row = m0 + r + 8 * hi, col = n0 + j * 16 + cl;
      float v = scale * c[j].f[r] + ((row == col) ? diag : 0.f);
      if (OUTn) OUTn[((size_t)bh << 16) + (size_t)row * 256 + col] = (_Float16)v;
      if (OUTt) OUTt[((size_t)bh << 16) + (size_t)col * 256 + row] = (_Float16)v;
    }
}

// ---------------- sim3 = q_l @ k^T -> f32 [bh][256][8192], 32x64 tile/wave ----------
__global__ void k_gemm_sim3(const _Float16* __restrict__ ql, const _Float16* __restrict__ kh,
                            float* __restrict__ sim3) {
  int lane = threadIdx.x & 31;
  int wav  = blockIdx.x * (blockDim.x >> 5) + (threadIdx.x >> 5);   // 16384 waves
  int bh = wav >> 10, t = wav & 1023;
  int mt = t >> 7, nt = t & 127;
  int m0 = mt << 5, n0 = nt << 6;
  const _Float16* Ab = ql + (size_t)bh * LMARK * DHEAD + (size_t)m0 * DHEAD;
  const _Float16* Bb = kh + (size_t)bh * SEQN * DHEAD;
  F32Acc c[2][4];
  v8f zero = {0.f, 0.f, 0.f, 0.f, 0.f, 0.f, 0.f, 0.f};
#pragma unroll
  for (int u = 0; u < 2; ++u)
#pragma unroll
    for (int j = 0; j < 4; ++j) c[u][j].v = zero;

  for (int kk = 0; kk < DHEAD; kk += 32) {
    v16h a0 = load_a(Ab + kk, DHEAD, lane);
    v16h a1 = load_a(Ab + (size_t)16 * DHEAD + kk, DHEAD, lane);
#pragma unroll
    for (int j = 0; j < 4; ++j) {
      v16h b = load_bt(Bb + (size_t)(n0 + j * 16) * DHEAD + kk, DHEAD, lane);
      c[0][j].v = wmma_f16(a0, b, c[0][j].v);
      c[1][j].v = wmma_f16(a1, b, c[1][j].v);
    }
  }
  int hi = lane >> 4, cl = lane & 15;
#pragma unroll
  for (int u = 0; u < 2; ++u)
#pragma unroll
    for (int j = 0; j < 4; ++j)
#pragma unroll
      for (int r = 0; r < 8; ++r) {
        int row = m0 + u * 16 + r + 8 * hi, col = n0 + j * 16 + cl;
        sim3[(size_t)bh * LMARK * SEQN + (size_t)row * SEQN + col] = c[u][j].f[r];
      }
}

// ---------------- row softmax over 8192 columns (LDS tree reduce) ----------------
__global__ void k_softmax_n(const float* __restrict__ S, _Float16* __restrict__ OUT) {
  __shared__ float red[256];
  int row = blockIdx.x;            // 4096 rows
  int tid = threadIdx.x;           // 256 threads
  const float* p = S + (size_t)row * SEQN;
  float vals[32];
  float mx = -1e30f;
#pragma unroll
  for (int i = 0; i < 32; ++i) { vals[i] = p[tid + (i << 8)]; mx = fmaxf(mx, vals[i]); }
  red[tid] = mx; __syncthreads();
  for (int s = 128; s > 0; s >>= 1) { if (tid < s) red[tid] = fmaxf(red[tid], red[tid + s]); __syncthreads(); }
  mx = red[0]; __syncthreads();
  float sum = 0.f;
#pragma unroll
  for (int i = 0; i < 32; ++i) { vals[i] = __expf(vals[i] - mx); sum += vals[i]; }
  red[tid] = sum; __syncthreads();
  for (int s = 128; s > 0; s >>= 1) { if (tid < s) red[tid] += red[tid + s]; __syncthreads(); }
  float inv = 1.f / red[0];
  _Float16* o = OUT + (size_t)row * SEQN;
#pragma unroll
  for (int i = 0; i < 32; ++i) o[tid + (i << 8)] = (_Float16)(vals[i] * inv);
}

// ---------------- v -> v^T per (b,h): [8192][64] -> [64][8192] ----------------
__global__ void k_transpose_v(const _Float16* __restrict__ vh, _Float16* __restrict__ vt) {
  const int n = BH * SEQN * DHEAD;
  for (int i = blockIdx.x * blockDim.x + threadIdx.x; i < n; i += gridDim.x * blockDim.x) {
    int bh = i / (SEQN * DHEAD);
    int rem = i - bh * (SEQN * DHEAD);
    int nn = rem >> 6, dd = rem & 63;
    vt[((size_t)bh * DHEAD + dd) * SEQN + nn] = vh[i];
  }
}

// ---------------- kv partials: split-K over n=8192 -> f32 [sp][bh][64][256] --------
__global__ void k_gemm_kv(const _Float16* __restrict__ attn3, const _Float16* __restrict__ vt,
                          float* __restrict__ kvp) {
  int lane = threadIdx.x & 31;
  int wav  = blockIdx.x * (blockDim.x >> 5) + (threadIdx.x >> 5);   // 4096 waves
  int sp = wav & (KSPLIT - 1);
  int mt = (wav >> 4) & 15;
  int bh = wav >> 8;
  int m0 = mt << 4;
  int k0 = sp * (SEQN / KSPLIT);                                    // 512-wide chunk
  const _Float16* Ab = attn3 + (size_t)bh * LMARK * SEQN + (size_t)m0 * SEQN;
  const _Float16* Bb = vt + (size_t)bh * DHEAD * SEQN;
  F32Acc c[4];
  v8f zero = {0.f, 0.f, 0.f, 0.f, 0.f, 0.f, 0.f, 0.f};
#pragma unroll
  for (int j = 0; j < 4; ++j) c[j].v = zero;

  for (int kk = k0; kk < k0 + SEQN / KSPLIT; kk += 32) {
    __builtin_prefetch(Ab + kk + 256, 0, 1);
    v16h a = load_a(Ab + kk, SEQN, lane);
#pragma unroll
    for (int j = 0; j < 4; ++j) {
      v16h b = load_bt(Bb + (size_t)(j * 16) * SEQN + kk, SEQN, lane);
      c[j].v = wmma_f16(a, b, c[j].v);
    }
  }
  int hi = lane >> 4, cl = lane & 15;
#pragma unroll
  for (int j = 0; j < 4; ++j)
#pragma unroll
    for (int r = 0; r < 8; ++r) {
      int row = m0 + r + 8 * hi;          // landmark index
      int col = j * 16 + cl;              // d index
      kvp[(((size_t)sp * BH + bh) * DHEAD + col) * LMARK + row] = c[j].f[r];
    }
}

// reduce split-K partials -> kv^T f16 [bh][64][256]
__global__ void k_kv_reduce(const float* __restrict__ kvp, _Float16* __restrict__ kvT) {
  const int n = BH * DHEAD * LMARK;
  for (int i = blockIdx.x * blockDim.x + threadIdx.x; i < n; i += gridDim.x * blockDim.x) {
    float s = 0.f;
#pragma unroll
    for (int sp = 0; sp < KSPLIT; ++sp) s += kvp[(size_t)sp * n + i];
    kvT[i] = (_Float16)s;
  }
}

// ---------------- w1 = attn1 @ attn2_inv -> f16 [bh][8192][256], 32x64 tile ---------
__global__ void k_gemm_w1(const _Float16* __restrict__ attn1, const _Float16* __restrict__ zT,
                          _Float16* __restrict__ w1) {
  int lane = threadIdx.x & 31;
  int wav  = blockIdx.x * (blockDim.x >> 5) + (threadIdx.x >> 5);   // 16384 waves
  int bh = wav >> 10, t = wav & 1023;
  int mt = t >> 2, nt = t & 3;
  int m0 = mt << 5, n0 = nt << 6;
  const _Float16* Ab = attn1 + (size_t)bh * SEQN * LMARK + (size_t)m0 * LMARK;
  const _Float16* Bb = zT + ((size_t)bh << 16);
  F32Acc c[2][4];
  v8f zero = {0.f, 0.f, 0.f, 0.f, 0.f, 0.f, 0.f, 0.f};
#pragma unroll
  for (int u = 0; u < 2; ++u)
#pragma unroll
    for (int j = 0; j < 4; ++j) c[u][j].v = zero;

  for (int kk = 0; kk < LMARK; kk += 32) {
    v16h a0 = load_a(Ab + kk, LMARK, lane);
    v16h a1 = load_a(Ab + (size_t)16 * LMARK + kk, LMARK, lane);
#pragma unroll
    for (int j = 0; j < 4; ++j) {
      v16h b = load_bt(Bb + (size_t)(n0 + j * 16) * 256 + kk, 256, lane);
      c[0][j].v = wmma_f16(a0, b, c[0][j].v);
      c[1][j].v = wmma_f16(a1, b, c[1][j].v);
    }
  }
  int hi = lane >> 4, cl = lane & 15;
#pragma unroll
  for (int u = 0; u < 2; ++u)
#pragma unroll
    for (int j = 0; j < 4; ++j)
#pragma unroll
      for (int r = 0; r < 8; ++r) {
        int row = m0 + u * 16 + r + 8 * hi, col = n0 + j * 16 + cl;
        w1[(size_t)bh * SEQN * LMARK + (size_t)row * LMARK + col] = (_Float16)c[u][j].f[r];
      }
}

// ---------------- out_heads = w1 @ kv + depthwise conv(v), store [b][n][h*d] f16 ----
__global__ void k_gemm_oh(const _Float16* __restrict__ w1, const _Float16* __restrict__ kvT,
                          const _Float16* __restrict__ vh, const float* __restrict__ Wres,
                          _Float16* __restrict__ oh) {
  int lane = threadIdx.x & 31;
  int wav  = blockIdx.x * (blockDim.x >> 5) + (threadIdx.x >> 5);   // 8192 waves
  int bh = wav >> 9, mt = wav & 511;
  int m0 = mt << 4;
  const _Float16* Ab = w1 + (size_t)bh * SEQN * LMARK + (size_t)m0 * LMARK;
  const _Float16* Bb = kvT + (size_t)bh * DHEAD * LMARK;
  F32Acc c[4];
  v8f zero = {0.f, 0.f, 0.f, 0.f, 0.f, 0.f, 0.f, 0.f};
#pragma unroll
  for (int j = 0; j < 4; ++j) c[j].v = zero;

  for (int kk = 0; kk < LMARK; kk += 32) {
    v16h a = load_a(Ab + kk, LMARK, lane);
#pragma unroll
    for (int j = 0; j < 4; ++j) {
      v16h b = load_bt(Bb + (size_t)(j * 16) * LMARK + kk, LMARK, lane);
      c[j].v = wmma_f16(a, b, c[j].v);
    }
  }

  int hi = lane >> 4, cl = lane & 15;
  int bb = bh >> 3, h = bh & 7;
  float wr[KCONV];
#pragma unroll
  for (int t = 0; t < KCONV; ++t) wr[t] = Wres[h * KCONV + t];
  const _Float16* vbase = vh + (size_t)bh * SEQN * DHEAD;

#pragma unroll
  for (int j = 0; j < 4; ++j) {
    int dd = j * 16 + cl;
    float vv[40];
    int start = m0 - 16 + 8 * hi;
#pragma unroll
    for (int s = 0; s < 40; ++s) {
      int nn = start + s;
      vv[s] = (nn >= 0 && nn < SEQN) ? (float)vbase[(size_t)nn * DHEAD + dd] : 0.f;
    }
#pragma unroll
    for (int r = 0; r < 8; ++r) {
      int nn = m0 + r + 8 * hi;
      float acc = c[j].f[r];
#pragma unroll
      for (int t = 0; t < KCONV; ++t) acc += wr[t] * vv[r + t];   // correlation, pad 16
      oh[((size_t)bb * SEQN + nn) * INNER + h * DHEAD + dd] = (_Float16)acc;
    }
  }
}

// ---------------- final: out = oh[16384,512] @ W_out + b_out (f32), 32x64 tile ------
__global__ void k_gemm_final(const _Float16* __restrict__ oh, const _Float16* __restrict__ woT,
                             const float* __restrict__ bout, float* __restrict__ out) {
  int lane = threadIdx.x & 31;
  int wav  = blockIdx.x * (blockDim.x >> 5) + (threadIdx.x >> 5);   // 4096 waves
  int mt = wav >> 3, nt = wav & 7;
  int m0 = mt << 5, n0 = nt << 6;
  const _Float16* Ab = oh + (size_t)m0 * INNER;
  F32Acc c[2][4];
  v8f zero = {0.f, 0.f, 0.f, 0.f, 0.f, 0.f, 0.f, 0.f};
#pragma unroll
  for (int u = 0; u < 2; ++u)
#pragma unroll
    for (int j = 0; j < 4; ++j) c[u][j].v = zero;

  for (int kk = 0; kk < INNER; kk += 32) {
    v16h a0 = load_a(Ab + kk, INNER, lane);
    v16h a1 = load_a(Ab + (size_t)16 * INNER + kk, INNER, lane);
#pragma unroll
    for (int j = 0; j < 4; ++j) {
      v16h b = load_bt(woT + (size_t)(n0 + j * 16) * INNER + kk, INNER, lane);
      c[0][j].v = wmma_f16(a0, b, c[0][j].v);
      c[1][j].v = wmma_f16(a1, b, c[1][j].v);
    }
  }
  int hi = lane >> 4, cl = lane & 15;
#pragma unroll
  for (int u = 0; u < 2; ++u)
#pragma unroll
    for (int j = 0; j < 4; ++j)
#pragma unroll
      for (int r = 0; r < 8; ++r) {
        int row = m0 + u * 16 + r + 8 * hi, col = n0 + j * 16 + cl;
        out[(size_t)row * DIM + col] = c[u][j].f[r] + bout[col];
      }
}

// ---------------- host orchestration ----------------
extern "C" void kernel_launch(void* const* d_in, const int* in_sizes, int n_in,
                              void* d_out, int out_size, void* d_ws, size_t ws_size,
                              hipStream_t stream) {
  (void)in_sizes; (void)n_in; (void)out_size; (void)ws_size;
  const float* x    = (const float*)d_in[0];
  const float* Wqkv = (const float*)d_in[1];
  const float* Wres = (const float*)d_in[2];
  const float* Wout = (const float*)d_in[3];
  const float* bout = (const float*)d_in[4];
  float* out = (float*)d_out;

  char* ws = (char*)d_ws;
  size_t off = 0;
  auto alloc = [&](size_t bytes) -> char* {
    char* p = ws + off;
    off = (off + bytes + 255) & ~(size_t)255;
    return p;
  };

  _Float16* xh    = (_Float16*)alloc((size_t)ROWS * DIM * 2);
  _Float16* wqt   = (_Float16*)alloc((size_t)QKVC * DIM * 2);
  _Float16* wot   = (_Float16*)alloc((size_t)DIM * INNER * 2);
  _Float16* qh    = (_Float16*)alloc((size_t)BH * SEQN * DHEAD * 2);
  _Float16* kh    = (_Float16*)alloc((size_t)BH * SEQN * DHEAD * 2);
  _Float16* vh    = (_Float16*)alloc((size_t)BH * SEQN * DHEAD * 2);
  _Float16* vt    = (_Float16*)alloc((size_t)BH * SEQN * DHEAD * 2);
  _Float16* qlh   = (_Float16*)alloc((size_t)BH * LMARK * DHEAD * 2);
  _Float16* klh   = (_Float16*)alloc((size_t)BH * LMARK * DHEAD * 2);
  _Float16* attn1 = (_Float16*)alloc((size_t)BH * SEQN * LMARK * 2);
  _Float16* a2h   = (_Float16*)alloc((size_t)BH * LMARK * LMARK * 2);
  _Float16* z0    = (_Float16*)alloc((size_t)BH * LMARK * LMARK * 2);
  _Float16* z0T   = (_Float16*)alloc((size_t)BH * LMARK * LMARK * 2);
  _Float16* z1    = (_Float16*)alloc((size_t)BH * LMARK * LMARK * 2);
  _Float16* z1T   = (_Float16*)alloc((size_t)BH * LMARK * LMARK * 2);
  _Float16* xz    = (_Float16*)alloc((size_t)BH * LMARK * LMARK * 2);
  _Float16* pbT   = (_Float16*)alloc((size_t)BH * LMARK * LMARK * 2);
  _Float16* qbT   = (_Float16*)alloc((size_t)BH * LMARK * LMARK * 2);
  unsigned* scale = (unsigned*)alloc(256);
  float*    kvp   = (float*)alloc((size_t)KSPLIT * BH * DHEAD * LMARK * 4);
  float*    sim3  = (float*)alloc((size_t)BH * LMARK * SEQN * 4);
  _Float16* attn3 = (_Float16*)alloc((size_t)BH * LMARK * SEQN * 2);
  _Float16* kvT   = (_Float16*)alloc((size_t)BH * DHEAD * LMARK * 2);
  _Float16* w1    = (_Float16*)sim3;   // alias: sim3 consumed before w1 written
  _Float16* oh    = xh;                // alias: xh dead after QKV projection

  const int TB = 256;
  k_cvt_half<<<2048, TB, 0, stream>>>(x, xh, ROWS * DIM);
  k_prep_wqkv<<<1024, TB, 0, stream>>>(Wqkv, wqt);
  k_prep_wout<<<512, TB, 0, stream>>>(Wout, wot);

  k_gemm_qkv<<<3072, 128, 0, stream>>>(xh, wqt, qh, kh, vh);       // 12288 waves
  k_landmarks<<<512, TB, 0, stream>>>(qh, kh, qlh, klh);

  k_qk_softmax<<<2048, 128, 0, stream>>>(qh, klh, attn1, SEQN);    // attn1 [bh,n,m]
  k_qk_softmax<<<64, 128, 0, stream>>>(qlh, klh, a2h, LMARK);      // attn2 [bh,m,m]

  // Moore-Penrose pinv of attn2 (Newton-Schulz, 6 iters), all WMMA, all b128 operands.
  k_zero_u32<<<1, 32, 0, stream>>>(scale);
  k_colsum_max<<<16, 256, 0, stream>>>(a2h, scale);
  k_init_z<<<1024, TB, 0, stream>>>(a2h, z0, z0T, scale);
  _Float16 *cur = z0, *curT = z0T, *nxt = z1, *nxtT = z1T;
  for (int it = 0; it < PINV_ITERS; ++it) {
    k_gemm256<<<256, 128, 0, stream>>>(a2h, curT, xz, nullptr, 0.f, 1.f);   // xz = x@z
    k_7i_minus_T<<<1024, TB, 0, stream>>>(xz, pbT);                         // p = 7I - xz
    k_gemm256<<<256, 128, 0, stream>>>(xz, pbT, nullptr, qbT, 15.f, -1.f);  // q = 15I - xz@p
    k_gemm256<<<256, 128, 0, stream>>>(xz, qbT, nullptr, pbT, 13.f, -1.f);  // p = 13I - xz@q
    k_gemm256<<<256, 128, 0, stream>>>(cur, pbT, nxt, nxtT, 0.f, 0.25f);    // z' = 0.25 z@p
    _Float16* t;
    t = cur; cur = nxt; nxt = t;
    t = curT; curT = nxtT; nxtT = t;
  }

  k_gemm_sim3<<<4096, 128, 0, stream>>>(qlh, kh, sim3);            // [bh,m,n] f32
  k_softmax_n<<<4096, 256, 0, stream>>>(sim3, attn3);              // attn3 f16
  k_transpose_v<<<2048, TB, 0, stream>>>(vh, vt);
  k_gemm_kv<<<1024, 128, 0, stream>>>(attn3, vt, kvp);             // split-K partials
  k_kv_reduce<<<256, TB, 0, stream>>>(kvp, kvT);                   // (attn3@v)^T f16
  k_gemm_w1<<<4096, 128, 0, stream>>>(attn1, curT, w1);            // attn1@pinv
  k_gemm_oh<<<2048, 128, 0, stream>>>(w1, kvT, vh, Wres, oh);      // + conv residual
  k_gemm_final<<<1024, 128, 0, stream>>>(oh, wot, bout, out);      // @W_out + b_out
}
